// MultiHeadAttention_9251359555869
// MI455X (gfx1250) — matrix operand order repaired and checked by reading, not compiled, not run on hardware
//
#include <hip/hip_runtime.h>
#include <hip/hip_bf16.h>

// ---------------------------------------------------------------------------
// MI455X (gfx1250) multi-head attention
//   - bf16 WMMA (v_wmma_f32_16x16x32_bf16) for all GEMMs, fp32 accumulate
//   - B-operand tiles double-buffered in LDS via global_load_async_to_lds_b128
//     (ASYNCcnt) when available; ds_load_b128 fragment reads
// ---------------------------------------------------------------------------

typedef __attribute__((ext_vector_type(16))) __bf16        v16bf;
typedef __attribute__((ext_vector_type(8)))  float         v8f;
typedef __attribute__((ext_vector_type(4)))  unsigned int  u32x4;
typedef __attribute__((ext_vector_type(4)))  float         f32x4;
typedef int v4i __attribute__((vector_size(16)));

union Frag16 { v16bf bf; u32x4 u[2]; };

#define D_MODEL 1024
#define SEQ     2048
#define BATCH   4
#define HEADS   16
#define DK      64
#define NEGV    (-1000000000.0f)

// ---------------------------------------------------------------------------
// Async global->LDS copy (16B per call), guarded per toolchain
// ---------------------------------------------------------------------------
#if __has_builtin(__builtin_amdgcn_global_load_async_to_lds_b128) && \
    __has_builtin(__builtin_amdgcn_s_wait_asynccnt)
#define HAS_ASYNC_LDS 1
#endif

#ifdef HAS_ASYNC_LDS
__device__ __forceinline__ void async_cp16(const unsigned short* g,
                                           unsigned short* l) {
  __builtin_amdgcn_global_load_async_to_lds_b128(
      (__attribute__((address_space(1))) v4i*)(uintptr_t)g,
      (__attribute__((address_space(3))) v4i*)(unsigned int)(uintptr_t)l,
      0, 0);
}
#define ASYNC_WAIT() __builtin_amdgcn_s_wait_asynccnt(0)
#else
__device__ __forceinline__ void async_cp16(const unsigned short* g,
                                           unsigned short* l) {
  *(u32x4*)l = *(const u32x4*)g;  // load + ds_store fallback
}
#define ASYNC_WAIT() ((void)0)
#endif

// Stage one 64col x 32k bf16 B-tile (4KB): 256 threads x 16B
// LDS layout: [col][kk] with kk contiguous (32 bf16 per col row)
__device__ __forceinline__ void stage_btile(const unsigned short* __restrict__ B,
                                            size_t stride, int col0, int k0,
                                            unsigned short* lbuf, int tid) {
  const int col = tid >> 2;
  const int kk  = (tid & 3) * 8;
  async_cp16(B + (size_t)(col0 + col) * stride + k0 + kk, lbuf + col * 32 + kk);
}

// round-to-nearest-even fp32 -> bf16 pack helpers
__device__ __forceinline__ unsigned int pk_bf16(float lo, float hi) {
  unsigned int l = __float_as_uint(lo);
  unsigned int h = __float_as_uint(hi);
  l += 0x7fffu + ((l >> 16) & 1u);
  h += 0x7fffu + ((h >> 16) & 1u);
  return (l >> 16) | (h & 0xffff0000u);
}
__device__ __forceinline__ unsigned short f2bf(float f) {
  unsigned int u = __float_as_uint(f);
  u += 0x7fffu + ((u >> 16) & 1u);
  return (unsigned short)(u >> 16);
}

// ---------------------------------------------------------------------------
// K0: fp32 -> bf16 bulk convert (n divisible by 4)
// ---------------------------------------------------------------------------
__global__ __launch_bounds__(256) void cvt_f32_bf16(
    const float* __restrict__ in, unsigned short* __restrict__ out, int n) {
  int i = (blockIdx.x * 256 + threadIdx.x) * 4;
  if (i + 3 < n) {
    f32x4 v = *(const f32x4*)(in + i);
    *(unsigned int*)(out + i)     = pk_bf16(v.x, v.y);
    *(unsigned int*)(out + i + 2) = pk_bf16(v.z, v.w);
  }
}

// ---------------------------------------------------------------------------
// K1/K5: GEMM  C[M,N] = A[M,K] @ W[N,K]^T + bias   (A,W bf16; acc fp32)
// wave tile 16x64, block = 8 waves -> 128x64; B tile staged in LDS (async,
// double buffered)
// MODE 0: fp32 flat [M,N]              MODE 1: bf16 [B,H,S,DK]
// MODE 2: bf16 [B,H,DK,S]
// ---------------------------------------------------------------------------
template <int MODE>
__global__ __launch_bounds__(256) void gemm_bf16(
    const unsigned short* __restrict__ A, const unsigned short* __restrict__ W,
    const float* __restrict__ bias, void* __restrict__ out, int N, int K) {
  __shared__ __align__(16) unsigned short btile[2][64 * 32];
  const int tid  = threadIdx.x;
  const int lane = tid & 31;
  const int wave = tid >> 5;
  const int lm   = lane & 15;
  const int g    = lane >> 4;
  const int rowb = blockIdx.y * 128 + wave * 16;
  const int row  = rowb + lm;
  const int col0 = blockIdx.x * 64;

  v8f acc[4] = {v8f{0.f}, v8f{0.f}, v8f{0.f}, v8f{0.f}};
  const unsigned short* Ap = A + (size_t)row * K;

  stage_btile(W, K, col0, 0, &btile[0][0], tid);
  const int nk = K / 32;
  for (int it = 0; it < nk; ++it) {
    const int k0 = it * 32;
    ASYNC_WAIT();
    __syncthreads();
    if (it + 1 < nk)
      stage_btile(W, K, col0, k0 + 32, &btile[(it + 1) & 1][0], tid);

    Frag16 a;
    a.u[0] = *(const u32x4*)(Ap + k0 + 8 * g);
    a.u[1] = *(const u32x4*)(Ap + k0 + 16 + 8 * g);
    const unsigned short* bt = &btile[it & 1][0];
#pragma unroll
    for (int j = 0; j < 4; ++j) {
      Frag16 b;
      const unsigned short* lp = bt + (j * 16 + lm) * 32 + 8 * g;
      b.u[0] = *(const u32x4*)(lp);
      b.u[1] = *(const u32x4*)(lp + 16);
      acc[j] = __builtin_amdgcn_wmma_f32_16x16x32_bf16(
          false, a.bf, false, b.bf, (short)0, acc[j], false, false);
    }
  }

#pragma unroll
  for (int j = 0; j < 4; ++j) {
    const int n  = col0 + j * 16 + lm;
    const float bv = bias[n];
#pragma unroll
    for (int r = 0; r < 8; ++r) {
      const int m = rowb + r + 8 * g;
      const float val = acc[j][r] + bv;
      if (MODE == 0) {
        ((float*)out)[(size_t)m * N + n] = val;
      } else {
        const int b = m >> 11, s = m & (SEQ - 1);
        const int h = n >> 6,  d = n & (DK - 1);
        size_t idx;
        if (MODE == 1)
          idx = (((size_t)(b * HEADS + h) * SEQ) + s) * DK + d;   // [B,H,S,DK]
        else
          idx = (((size_t)(b * HEADS + h) * DK) + d) * SEQ + s;   // [B,H,DK,S]
        ((unsigned short*)out)[idx] = f2bf(val);
      }
    }
  }
}

// ---------------------------------------------------------------------------
// K2: scores[z,q,k] = (Qh . Kh)/8, masked -> fp32 into d_out attn region
// z = b*16+h ; M=N=SEQ, K=DK=64 ; K-tile staged in LDS
// ---------------------------------------------------------------------------
__global__ __launch_bounds__(256) void attn_scores(
    const unsigned short* __restrict__ Qh, const unsigned short* __restrict__ Kh,
    const int* __restrict__ mask, float* __restrict__ attn) {
  __shared__ __align__(16) unsigned short btile[2][64 * 32];
  const int z = blockIdx.z;
  const unsigned short* Qz = Qh + (size_t)z * SEQ * DK;
  const unsigned short* Kz = Kh + (size_t)z * SEQ * DK;
  const int tid = threadIdx.x;
  const int lane = tid & 31;
  const int wave = tid >> 5;
  const int lm = lane & 15, g = lane >> 4;
  const int rowb = blockIdx.y * 128 + wave * 16;
  const int row  = rowb + lm;
  const int col0 = blockIdx.x * 64;

  v8f acc[4] = {v8f{0.f}, v8f{0.f}, v8f{0.f}, v8f{0.f}};
  const unsigned short* Ap = Qz + (size_t)row * DK;

  stage_btile(Kz, DK, col0, 0, &btile[0][0], tid);
#pragma unroll
  for (int it = 0; it < 2; ++it) {
    const int k0 = it * 32;
    ASYNC_WAIT();
    __syncthreads();
    if (it + 1 < 2) stage_btile(Kz, DK, col0, 32, &btile[1][0], tid);

    Frag16 a;
    a.u[0] = *(const u32x4*)(Ap + k0 + 8 * g);
    a.u[1] = *(const u32x4*)(Ap + k0 + 16 + 8 * g);
    const unsigned short* bt = &btile[it & 1][0];
#pragma unroll
    for (int j = 0; j < 4; ++j) {
      Frag16 b;
      const unsigned short* lp = bt + (j * 16 + lm) * 32 + 8 * g;
      b.u[0] = *(const u32x4*)(lp);
      b.u[1] = *(const u32x4*)(lp + 16);
      acc[j] = __builtin_amdgcn_wmma_f32_16x16x32_bf16(
          false, a.bf, false, b.bf, (short)0, acc[j], false, false);
    }
  }

  const int b = z >> 4;
#pragma unroll
  for (int j = 0; j < 4; ++j) {
    const int n = col0 + j * 16 + lm;
#pragma unroll
    for (int r = 0; r < 8; ++r) {
      const int m = rowb + r + 8 * g;
      const float v = acc[j][r] * 0.125f;  // 1/sqrt(64)
      const int mk = mask[((size_t)b * SEQ + m) * SEQ + n];
      attn[((size_t)z * SEQ + m) * SEQ + n] = mk ? v : NEGV;
    }
  }
}

// ---------------------------------------------------------------------------
// K3: in-place row softmax over length-2048 rows (block per row)
// ---------------------------------------------------------------------------
__global__ __launch_bounds__(256) void softmax_rows(float* __restrict__ attn) {
  float* p = attn + (size_t)blockIdx.x * SEQ;
  const int tid = threadIdx.x;
  __shared__ float sm[256];

  float vals[8];
  float mx = -3.4e38f;
#pragma unroll
  for (int i = 0; i < 8; ++i) {
    vals[i] = p[tid + i * 256];
    mx = fmaxf(mx, vals[i]);
  }
  sm[tid] = mx;
  __syncthreads();
  for (int off = 128; off > 0; off >>= 1) {
    if (tid < off) sm[tid] = fmaxf(sm[tid], sm[tid + off]);
    __syncthreads();
  }
  const float rowmax = sm[0];
  __syncthreads();

  float s = 0.f;
#pragma unroll
  for (int i = 0; i < 8; ++i) {
    vals[i] = __expf(vals[i] - rowmax);
    s += vals[i];
  }
  sm[tid] = s;
  __syncthreads();
  for (int off = 128; off > 0; off >>= 1) {
    if (tid < off) sm[tid] += sm[tid + off];
    __syncthreads();
  }
  const float inv = 1.f / sm[0];
#pragma unroll
  for (int i = 0; i < 8; ++i) p[tid + i * 256] = vals[i] * inv;
}

// ---------------------------------------------------------------------------
// K4: out[z,q,d] = attn[z,q,:] @ Vt[z,d,:]^T ; write bf16 combined [B,S,1024]
// A packed on the fly fp32->bf16; B = transposed V staged in LDS
// ---------------------------------------------------------------------------
__global__ __launch_bounds__(256) void attn_v(
    const float* __restrict__ attn, const unsigned short* __restrict__ Vt,
    unsigned short* __restrict__ Cb) {
  __shared__ __align__(16) unsigned short btile[2][64 * 32];
  const int z = blockIdx.z;
  const float* Az = attn + (size_t)z * SEQ * SEQ;
  const unsigned short* Vz = Vt + (size_t)z * DK * SEQ;
  const int tid = threadIdx.x;
  const int lane = tid & 31;
  const int wave = tid >> 5;
  const int lm = lane & 15, g = lane >> 4;
  const int rowb = blockIdx.y * 128 + wave * 16;
  const int row  = rowb + lm;

  v8f acc[4] = {v8f{0.f}, v8f{0.f}, v8f{0.f}, v8f{0.f}};
  const float* Ap = Az + (size_t)row * SEQ;

  stage_btile(Vz, SEQ, 0, 0, &btile[0][0], tid);
  const int nk = SEQ / 32;
  for (int it = 0; it < nk; ++it) {
    const int k0 = it * 32;
    ASYNC_WAIT();
    __syncthreads();
    if (it + 1 < nk) stage_btile(Vz, SEQ, 0, k0 + 32, &btile[(it + 1) & 1][0], tid);

    f32x4 a0 = *(const f32x4*)(Ap + k0 + 8 * g);
    f32x4 a1 = *(const f32x4*)(Ap + k0 + 8 * g + 4);
    f32x4 a2 = *(const f32x4*)(Ap + k0 + 16 + 8 * g);
    f32x4 a3 = *(const f32x4*)(Ap + k0 + 16 + 8 * g + 4);
    Frag16 a;
    u32x4 t0, t1;
    t0.x = pk_bf16(a0.x, a0.y); t0.y = pk_bf16(a0.z, a0.w);
    t0.z = pk_bf16(a1.x, a1.y); t0.w = pk_bf16(a1.z, a1.w);
    t1.x = pk_bf16(a2.x, a2.y); t1.y = pk_bf16(a2.z, a2.w);
    t1.z = pk_bf16(a3.x, a3.y); t1.w = pk_bf16(a3.z, a3.w);
    a.u[0] = t0;
    a.u[1] = t1;

    const unsigned short* bt = &btile[it & 1][0];
#pragma unroll
    for (int j = 0; j < 4; ++j) {
      Frag16 b;
      const unsigned short* lp = bt + (j * 16 + lm) * 32 + 8 * g;
      b.u[0] = *(const u32x4*)(lp);
      b.u[1] = *(const u32x4*)(lp + 16);
      acc[j] = __builtin_amdgcn_wmma_f32_16x16x32_bf16(
          false, a.bf, false, b.bf, (short)0, acc[j], false, false);
    }
  }

  const int b = z >> 4, h = z & (HEADS - 1);
#pragma unroll
  for (int j = 0; j < 4; ++j) {
    const int d = j * 16 + lm;
#pragma unroll
    for (int r = 0; r < 8; ++r) {
      const int q = rowb + r + 8 * g;
      Cb[((size_t)b * SEQ + q) * D_MODEL + h * DK + d] = f2bf(acc[j][r]);
    }
  }
}

// ---------------------------------------------------------------------------
// Launch
// ---------------------------------------------------------------------------
extern "C" void kernel_launch(void* const* d_in, const int* in_sizes, int n_in,
                              void* d_out, int out_size, void* d_ws,
                              size_t ws_size, hipStream_t stream) {
  (void)in_sizes; (void)n_in; (void)out_size; (void)ws_size;
  // input order: q,k,v,mask, wq,bq, wk,bk, wv,bv, wo,bo
  const float* q    = (const float*)d_in[0];
  const float* k    = (const float*)d_in[1];
  const float* v    = (const float*)d_in[2];
  const int*   mask = (const int*)  d_in[3];
  const float* wq   = (const float*)d_in[4];
  const float* bq   = (const float*)d_in[5];
  const float* wk   = (const float*)d_in[6];
  const float* bk   = (const float*)d_in[7];
  const float* wv   = (const float*)d_in[8];
  const float* bv   = (const float*)d_in[9];
  const float* wo   = (const float*)d_in[10];
  const float* bo   = (const float*)d_in[11];

  const size_t SZ_ACT = (size_t)BATCH * SEQ * D_MODEL;  // 8.39M elems
  const size_t SZ_W   = (size_t)D_MODEL * D_MODEL;      // 1.05M elems

  unsigned short* ws  = (unsigned short*)d_ws;
  unsigned short* qb  = ws;
  unsigned short* kb  = qb  + SZ_ACT;
  unsigned short* vb  = kb  + SZ_ACT;
  unsigned short* wqb = vb  + SZ_ACT;
  unsigned short* wkb = wqb + SZ_W;
  unsigned short* wvb = wkb + SZ_W;
  unsigned short* wob = wvb + SZ_W;
  unsigned short* Qh  = wob + SZ_W;   // [B,H,S,DK] bf16
  unsigned short* Kh  = Qh  + SZ_ACT;
  unsigned short* Vt  = Kh  + SZ_ACT; // [B,H,DK,S] bf16
  unsigned short* Cb  = Vt  + SZ_ACT; // [B,S,D_MODEL] bf16

  float* out_proj = (float*)d_out;                 // [B,S,D_MODEL] fp32
  float* attn     = (float*)d_out + SZ_ACT;        // [B,H,S,S] fp32

  // K0: convert to bf16
  cvt_f32_bf16<<<(SZ_ACT / 4 + 255) / 256, 256, 0, stream>>>(q, qb, (int)SZ_ACT);
  cvt_f32_bf16<<<(SZ_ACT / 4 + 255) / 256, 256, 0, stream>>>(k, kb, (int)SZ_ACT);
  cvt_f32_bf16<<<(SZ_ACT / 4 + 255) / 256, 256, 0, stream>>>(v, vb, (int)SZ_ACT);
  cvt_f32_bf16<<<(SZ_W / 4 + 255) / 256, 256, 0, stream>>>(wq, wqb, (int)SZ_W);
  cvt_f32_bf16<<<(SZ_W / 4 + 255) / 256, 256, 0, stream>>>(wk, wkb, (int)SZ_W);
  cvt_f32_bf16<<<(SZ_W / 4 + 255) / 256, 256, 0, stream>>>(wv, wvb, (int)SZ_W);
  cvt_f32_bf16<<<(SZ_W / 4 + 255) / 256, 256, 0, stream>>>(wo, wob, (int)SZ_W);

  // K1: Q/K/V projections (M=8192 rows, N=K=1024)
  dim3 gproj(D_MODEL / 64, (BATCH * SEQ) / 128, 1);
  gemm_bf16<1><<<gproj, 256, 0, stream>>>(qb, wqb, bq, Qh, D_MODEL, D_MODEL);
  gemm_bf16<1><<<gproj, 256, 0, stream>>>(kb, wkb, bk, Kh, D_MODEL, D_MODEL);
  gemm_bf16<2><<<gproj, 256, 0, stream>>>(vb, wvb, bv, Vt, D_MODEL, D_MODEL);

  // K2: masked scaled scores -> fp32 attn region of d_out
  dim3 gsc(SEQ / 64, SEQ / 128, BATCH * HEADS);
  attn_scores<<<gsc, 256, 0, stream>>>(Qh, Kh, mask, attn);

  // K3: softmax in place
  softmax_rows<<<(unsigned)(BATCH * HEADS * SEQ), 256, 0, stream>>>(attn);

  // K4: attn @ V -> combined bf16
  dim3 gav(1, SEQ / 128, BATCH * HEADS);
  attn_v<<<gav, 256, 0, stream>>>(attn, Vt, Cb);

  // K5: output projection -> fp32 d_out
  gemm_bf16<0><<<gproj, 256, 0, stream>>>(Cb, wob, bo, out_proj, D_MODEL, D_MODEL);
}
